// GALE_block_74912819577589
// MI455X (gfx1250) — matrix operand order repaired and checked by reading, not compiled, not run on hardware
//
#include <hip/hip_runtime.h>
#include <hip/hip_bf16.h>

typedef __attribute__((ext_vector_type(16))) _Float16 v16h;
typedef __attribute__((ext_vector_type(8)))  float    v8f;

#define BB 2
#define NN 32768
#define CC 512
#define HH 8
#define DH 64
#define GG 64
#define CD 256

// ---------------------------------------------------------------------------
// CDNA5 async global->LDS copy (ASYNCcnt path, ISA §10 / §15.18.3).
// Per-lane: copies 16 bytes from lane's global addr to lane's LDS addr.
// Generic shared pointers carry the LDS byte offset in their low 32 bits.
// ---------------------------------------------------------------------------
__device__ __forceinline__ void async_ld_b128(void* lds, const void* g) {
  unsigned l = (unsigned)(uintptr_t)lds;
  asm volatile("global_load_async_to_lds_b128 %0, %1, off"
               :: "v"(l), "v"(g) : "memory");
}
__device__ __forceinline__ void async_wait() {
  asm volatile("s_wait_asynccnt 0" ::: "memory");
}

// ---------------------------------------------------------------------------
// Fragment helpers (wave32; A/B in this code's contiguous k order, C/D per CDNA5 ISA §7.12.2)
// A 16x32 f16: lanes 0-15 -> M=0..15 ; element e: K = (lane>>4)*16 + e
// B 32x16 f16: lane = N (mod 16); (vgpr,half) j: K = (lane>>4)*16 + j
// C/D 16x16 f32: reg r: M = r + 8*(lane>>4), N = lane&15
// ---------------------------------------------------------------------------
__device__ __forceinline__ v16h a_frag(const _Float16* A, int lda, int row0, int k0) {
  int lane = threadIdx.x & 31;
  int g = lane >> 4;
  const _Float16* base = A + (size_t)(row0 + (lane & 15)) * lda + k0;
  v16h a;
#pragma unroll
  for (int v = 0; v < 4; ++v) {
    int k = g * 16 + 2 * v;
    a[2 * v]     = base[k];
    a[2 * v + 1] = base[k + 1];
    int k2 = g * 16 + 8 + 2 * v;
    a[8 + 2 * v]     = base[k2];
    a[8 + 2 * v + 1] = base[k2 + 1];
  }
  return a;
}

__device__ __forceinline__ v16h b_frag_lds(const _Float16* Bm, int ldb, int k0, int n0) {
  int lane = threadIdx.x & 31;
  int n = n0 + (lane & 15);
  int kb = k0 + ((lane >> 4) << 4);
  v16h b;
#pragma unroll
  for (int j = 0; j < 16; ++j) b[j] = Bm[(size_t)(kb + j) * ldb + n];
  return b;
}

#define WMMA(a, b, c) __builtin_amdgcn_wmma_f32_16x16x32_f16(false, (a), false, (b), (short)0, (c), false, false)

// ---------------------------------------------------------------------------
// Pack f32 weight (K x Ncols, row-major) into B-fragment tiles of 32x16.
// ---------------------------------------------------------------------------
__global__ __launch_bounds__(256) void pack_b(const float* W, _Float16* Wp, int K, int Ncols) {
  int tiles = (K / 32) * (Ncols / 16);
  int idx = blockIdx.x * 256 + threadIdx.x;
  if (idx >= tiles * 32) return;
  int t = idx >> 5, lane = idx & 31;
  int ntiles = Ncols / 16;
  int kt = t / ntiles, nt = t % ntiles;
  int n = nt * 16 + (lane & 15);
  int kb = kt * 32 + ((lane >> 4) << 4);
  _Float16* dst = Wp + (size_t)t * 512 + lane * 16;
#pragma unroll
  for (int j = 0; j < 16; ++j) dst[j] = (_Float16)W[(size_t)(kb + j) * Ncols + n];
}

__global__ void zero_f32(float* a, int n) {
  int i = blockIdx.x * 256 + threadIdx.x;
  if (i < n) a[i] = 0.f;
}

// ---------------------------------------------------------------------------
// GEMM1: x (65536x512) @ {Wx,Wfx} -> x_mid, fx  as f16 in (B,H,N,64) layout.
// ---------------------------------------------------------------------------
__global__ __launch_bounds__(256) void gemm_x_dual(
    const float* __restrict__ x, const _Float16* __restrict__ Wxp,
    const _Float16* __restrict__ Wfxp, const float* __restrict__ bx,
    const float* __restrict__ bfx, _Float16* __restrict__ xmid,
    _Float16* __restrict__ fx) {
  __shared__ _Float16 As[128 * 32];
  int wave = threadIdx.x >> 5, lane = threadIdx.x & 31;
  int rowBlock = blockIdx.x * 128;
  int colBlock = blockIdx.y * 64;
  v8f acc1[4] = {}, acc2[4] = {};
  for (int kt = 0; kt < 16; ++kt) {
    int r = threadIdx.x >> 1;
    int ks = (threadIdx.x & 1) * 16;
    const float* src = x + (size_t)(rowBlock + r) * CC + kt * 32 + ks;
    if (kt < 15) __builtin_prefetch(src + 32, 0, 0);   // next A panel
    _Float16* dst = As + r * 32 + ks;
#pragma unroll
    for (int j = 0; j < 16; ++j) dst[j] = (_Float16)src[j];
    __syncthreads();
    v16h a = a_frag(As, 32, wave * 16, 0);
#pragma unroll
    for (int j = 0; j < 4; ++j) {
      size_t t = (size_t)kt * 32 + (colBlock >> 4) + j;
      v16h b1 = *(const v16h*)(Wxp + t * 512 + lane * 16);
      v16h b2 = *(const v16h*)(Wfxp + t * 512 + lane * 16);
      acc1[j] = WMMA(a, b1, acc1[j]);
      acc2[j] = WMMA(a, b2, acc2[j]);
    }
    __syncthreads();
  }
  int g = lane >> 4, nIdx = lane & 15;
#pragma unroll
  for (int j = 0; j < 4; ++j) {
    int c = colBlock + 16 * j + nIdx;
    int h = c >> 6, d = c & 63;
    float b1v = bx[c], b2v = bfx[c];
#pragma unroll
    for (int r = 0; r < 8; ++r) {
      int m = rowBlock + wave * 16 + g * 8 + r;
      int bI = m >> 15;
      int n = m & (NN - 1);
      size_t off = ((size_t)(bI * HH + h) * NN + n) * 64 + d;
      xmid[off] = (_Float16)(acc1[j][r] + b1v);
      fx[off]   = (_Float16)(acc2[j][r] + b2v);
    }
  }
}

// ---------------------------------------------------------------------------
// logits = x_mid @ Wslice, 1/clip(temp) scale, softmax over G=64 -> w (f16),
// norm accumulation via LDS ds_add_f32 then one global atomic per column.
// A tile staged with async global->LDS (ASYNCcnt).
// ---------------------------------------------------------------------------
__global__ __launch_bounds__(256) void slice_softmax(
    const _Float16* __restrict__ xmid, const _Float16* __restrict__ Wsp,
    const float* __restrict__ bslice, const float* __restrict__ temperature,
    _Float16* __restrict__ w, float* __restrict__ norm) {
  __shared__ _Float16 As[128 * 64];
  __shared__ float normS[64];
  int bh = blockIdx.y;
  int h = bh & (HH - 1);
  int rowBlock = blockIdx.x * 128;
  int wave = threadIdx.x >> 5, lane = threadIdx.x & 31;
  if (threadIdx.x < 64) normS[threadIdx.x] = 0.f;
  {
    const char* src = (const char*)(xmid + ((size_t)bh * NN + rowBlock) * 64);
    char* dst = (char*)As;
#pragma unroll
    for (int i = 0; i < 4; ++i) {
      int e = threadIdx.x + i * 256;          // 1024 x 16B = 16 KB tile
      async_ld_b128(dst + (size_t)e * 16, src + (size_t)e * 16);
    }
    async_wait();
  }
  __syncthreads();
  v8f acc[4] = {};
#pragma unroll
  for (int kt = 0; kt < 2; ++kt) {
    v16h a = a_frag(As, 64, wave * 16, kt * 32);
#pragma unroll
    for (int j = 0; j < 4; ++j) {
      v16h b = *(const v16h*)(Wsp + (size_t)(kt * 4 + j) * 512 + lane * 16);
      acc[j] = WMMA(a, b, acc[j]);
    }
  }
  float tv = temperature[h];
  tv = fminf(fmaxf(tv, 0.1f), 5.0f);
  float invT = 1.f / tv;
  int g = lane >> 4, nIdx = lane & 15;
#pragma unroll
  for (int j = 0; j < 4; ++j) {
    float bb = bslice[16 * j + nIdx];
#pragma unroll
    for (int r = 0; r < 8; ++r) acc[j][r] = (acc[j][r] + bb) * invT;
  }
#pragma unroll
  for (int r = 0; r < 8; ++r) {
    float m = -1e30f;
#pragma unroll
    for (int j = 0; j < 4; ++j) m = fmaxf(m, acc[j][r]);
    for (int off = 1; off < 16; off <<= 1) m = fmaxf(m, __shfl_xor(m, off, 32));
    float s = 0.f, e[4];
#pragma unroll
    for (int j = 0; j < 4; ++j) { e[j] = __expf(acc[j][r] - m); s += e[j]; }
    for (int off = 1; off < 16; off <<= 1) s += __shfl_xor(s, off, 32);
    float inv = 1.f / s;
#pragma unroll
    for (int j = 0; j < 4; ++j) acc[j][r] = e[j] * inv;
  }
#pragma unroll
  for (int j = 0; j < 4; ++j) {
    int c = 16 * j + nIdx;
    float colsum = 0.f;
#pragma unroll
    for (int r = 0; r < 8; ++r) {
      int n = rowBlock + wave * 16 + g * 8 + r;
      w[((size_t)bh * NN + n) * 64 + c] = (_Float16)acc[j][r];
      colsum += acc[j][r];
    }
    colsum += __shfl_xor(colsum, 16, 32);
    if (g == 0) atomicAdd(&normS[c], colsum);
  }
  __syncthreads();
  if (threadIdx.x < 64) atomicAdd(&norm[bh * 64 + threadIdx.x], normS[threadIdx.x]);
}

// ---------------------------------------------------------------------------
// tokens[b,h] = w^T (64 x N) @ fx (N x 64), split-K over 8 chunks, f32 atomics.
// Tile staging via async global->LDS.
// ---------------------------------------------------------------------------
__global__ __launch_bounds__(256) void tokens_gemm(
    const _Float16* __restrict__ w, const _Float16* __restrict__ fx,
    float* __restrict__ tokens) {
  __shared__ _Float16 Ws[32 * 64];
  __shared__ _Float16 Fs[32 * 64];
  int bh = blockIdx.y;
  const int CH = NN / 8;
  int wave = threadIdx.x >> 5, lane = threadIdx.x & 31;
  int mt = wave & 3;             // row tile (g dim)
  int ct0 = (wave >> 2) * 2;     // two col tiles
  v8f acc[2] = {};
  size_t baseN = (size_t)bh * NN + (size_t)blockIdx.x * CH;
  for (int ks = 0; ks < CH / 32; ++ks) {
    size_t byteOff = (size_t)threadIdx.x * 16;            // 32*64*2 = 4 KB each
    const char* sw = (const char*)(w + (baseN + ks * 32) * 64);
    const char* sf = (const char*)(fx + (baseN + ks * 32) * 64);
    async_ld_b128((char*)Ws + byteOff, sw + byteOff);
    async_ld_b128((char*)Fs + byteOff, sf + byteOff);
    async_wait();
    __syncthreads();
    // A = w^T : A[m=g][k=n] = Ws[k][m]
    v16h a;
    {
      int gg = lane >> 4;
      int row = 16 * mt + (lane & 15);
#pragma unroll
      for (int v = 0; v < 4; ++v) {
        int k = gg * 16 + 2 * v;
        a[2 * v]     = Ws[k * 64 + row];
        a[2 * v + 1] = Ws[(k + 1) * 64 + row];
        int k2 = gg * 16 + 8 + 2 * v;
        a[8 + 2 * v]     = Ws[k2 * 64 + row];
        a[8 + 2 * v + 1] = Ws[(k2 + 1) * 64 + row];
      }
    }
#pragma unroll
    for (int j = 0; j < 2; ++j) {
      v16h b = b_frag_lds(Fs, 64, 0, (ct0 + j) * 16);
      acc[j] = WMMA(a, b, acc[j]);
    }
    __syncthreads();
  }
  int gg = lane >> 4, nIdx = lane & 15;
#pragma unroll
  for (int j = 0; j < 2; ++j) {
    int c = (ct0 + j) * 16 + nIdx;
#pragma unroll
    for (int r = 0; r < 8; ++r) {
      int m = 16 * mt + gg * 8 + r;
      atomicAdd(&tokens[((size_t)bh * 64 + m) * 64 + c], acc[j][r]);
    }
  }
}

// ---------------------------------------------------------------------------
// Tiny per-(b,h) token attention (self + cross + sigmoid gate), f32 VALU.
// ---------------------------------------------------------------------------
__global__ __launch_bounds__(256) void attn_kernel(
    const float* __restrict__ tokens, const float* __restrict__ norm,
    const float* Wq, const float* bq, const float* Wk, const float* bk,
    const float* Wv, const float* bv, const float* Wcq, const float* bcq,
    const float* Wck, const float* bck, const float* Wcv, const float* bcv,
    const float* __restrict__ context, const float* __restrict__ state_mixing,
    float* __restrict__ scratch, _Float16* __restrict__ outTokP) {
  int bh = blockIdx.x;
  float* tok = scratch + (size_t)bh * 10 * 4096;
  float* q = tok + 4096;  float* k = tok + 2 * 4096;
  float* v = tok + 3 * 4096; float* s = tok + 4 * 4096;
  float* sa = tok + 5 * 4096; float* cq = tok + 6 * 4096;
  float* ck = tok + 7 * 4096; float* cv = tok + 8 * 4096;
  float* ca = tok + 9 * 4096;
  int t = threadIdx.x;
  for (int i = t; i < 4096; i += 256) {
    int m = i >> 6;
    tok[i] = tokens[(size_t)bh * 4096 + i] / (norm[bh * 64 + m] + 1e-5f);
  }
  __syncthreads();
  for (int i = t; i < 4096; i += 256) {
    int m = i >> 6, d = i & 63;
    float a1 = bq[d], a2 = bk[d], a3 = bv[d], a4 = bcq[d];
    for (int kk = 0; kk < 64; ++kk) {
      float xv = tok[m * 64 + kk];
      a1 += xv * Wq[kk * 64 + d];  a2 += xv * Wk[kk * 64 + d];
      a3 += xv * Wv[kk * 64 + d];  a4 += xv * Wcq[kk * 64 + d];
    }
    q[i] = a1; k[i] = a2; v[i] = a3; cq[i] = a4;
  }
  const float* ctx = context + (size_t)bh * 64 * CD;
  for (int i = t; i < 4096; i += 256) {
    int m = i >> 6, d = i & 63;
    float a1 = bck[d], a2 = bcv[d];
    for (int kk = 0; kk < CD; ++kk) {
      float xv = ctx[m * CD + kk];
      a1 += xv * Wck[kk * 64 + d];  a2 += xv * Wcv[kk * 64 + d];
    }
    ck[i] = a1; cv[i] = a2;
  }
  __syncthreads();
  for (int i = t; i < 4096; i += 256) {
    int r = i >> 6, c = i & 63;
    float a = 0.f;
    for (int kk = 0; kk < 64; ++kk) a += q[r * 64 + kk] * k[c * 64 + kk];
    s[i] = a * 0.125f;
  }
  __syncthreads();
  if (t < 64) {
    float m = -1e30f;
    for (int j = 0; j < 64; ++j) m = fmaxf(m, s[t * 64 + j]);
    float sum = 0.f;
    for (int j = 0; j < 64; ++j) { float e = __expf(s[t * 64 + j] - m); s[t * 64 + j] = e; sum += e; }
    float inv = 1.f / sum;
    for (int j = 0; j < 64; ++j) s[t * 64 + j] *= inv;
  }
  __syncthreads();
  for (int i = t; i < 4096; i += 256) {
    int r = i >> 6, d = i & 63;
    float a = 0.f;
    for (int j = 0; j < 64; ++j) a += s[r * 64 + j] * v[j * 64 + d];
    sa[i] = a;
  }
  __syncthreads();
  for (int i = t; i < 4096; i += 256) {
    int r = i >> 6, c = i & 63;
    float a = 0.f;
    for (int kk = 0; kk < 64; ++kk) a += cq[r * 64 + kk] * ck[c * 64 + kk];
    s[i] = a * 0.125f;
  }
  __syncthreads();
  if (t < 64) {
    float m = -1e30f;
    for (int j = 0; j < 64; ++j) m = fmaxf(m, s[t * 64 + j]);
    float sum = 0.f;
    for (int j = 0; j < 64; ++j) { float e = __expf(s[t * 64 + j] - m); s[t * 64 + j] = e; sum += e; }
    float inv = 1.f / sum;
    for (int j = 0; j < 64; ++j) s[t * 64 + j] *= inv;
  }
  __syncthreads();
  for (int i = t; i < 4096; i += 256) {
    int r = i >> 6, d = i & 63;
    float a = 0.f;
    for (int j = 0; j < 64; ++j) a += s[r * 64 + j] * cv[j * 64 + d];
    ca[i] = a;
  }
  __syncthreads();
  float gmix = 1.f / (1.f + __expf(-state_mixing[0]));
  for (int i = t; i < 4096; i += 256) tok[i] = gmix * sa[i] + (1.f - gmix) * ca[i];
  __syncthreads();
  int tile = t >> 5, lane = t & 31;
  int kt = tile >> 2, nt = tile & 3;
  int n = nt * 16 + (lane & 15);
  int kb = kt * 32 + ((lane >> 4) << 4);
  _Float16* dst = outTokP + ((size_t)bh * 8 + tile) * 512 + lane * 16;
#pragma unroll
  for (int j = 0; j < 16; ++j) dst[j] = (_Float16)tok[(kb + j) * 64 + n];
}

// ---------------------------------------------------------------------------
// out_x[b,h] = w (N x 64) @ out_tok (64 x 64), f16 output (reuses xmid buffer).
// ---------------------------------------------------------------------------
__global__ __launch_bounds__(256) void deslice_gemm(
    const _Float16* __restrict__ w, const _Float16* __restrict__ outTokP,
    _Float16* __restrict__ outx) {
  int bh = blockIdx.y;
  int rowBlock = blockIdx.x * 128;
  int wave = threadIdx.x >> 5, lane = threadIdx.x & 31;
  const _Float16* wbase = w + (size_t)bh * NN * 64;
  v8f acc[4] = {};
#pragma unroll
  for (int kt = 0; kt < 2; ++kt) {
    v16h a = a_frag(wbase, 64, rowBlock + wave * 16, kt * 32);
#pragma unroll
    for (int j = 0; j < 4; ++j) {
      v16h b = *(const v16h*)(outTokP + ((size_t)bh * 8 + kt * 4 + j) * 512 + lane * 16);
      acc[j] = WMMA(a, b, acc[j]);
    }
  }
  int g = lane >> 4, nIdx = lane & 15;
#pragma unroll
  for (int j = 0; j < 4; ++j) {
    int c = 16 * j + nIdx;
#pragma unroll
    for (int r = 0; r < 8; ++r) {
      int n = rowBlock + wave * 16 + g * 8 + r;
      outx[((size_t)bh * NN + n) * 64 + c] = (_Float16)acc[j][r];
    }
  }
}

// ---------------------------------------------------------------------------
// Final GEMM: out = out_x (re-laid as (B,N,512)) @ Wo + bo, f32 output.
// A tile staged with async global->LDS (f16 direct copy, 32 B per lane).
// ---------------------------------------------------------------------------
__global__ __launch_bounds__(256) void out_gemm(
    const _Float16* __restrict__ outx, const _Float16* __restrict__ Wop,
    const float* __restrict__ bo, float* __restrict__ out) {
  __shared__ _Float16 As[128 * 32];
  int wave = threadIdx.x >> 5, lane = threadIdx.x & 31;
  int rowBlock = blockIdx.x * 128;
  int colBlock = blockIdx.y * 64;
  v8f acc[4] = {};
  for (int kt = 0; kt < 16; ++kt) {
    int r = threadIdx.x >> 1;
    int ks = (threadIdx.x & 1) * 16;
    int m = rowBlock + r, bI = m >> 15, n = m & (NN - 1);
    int c0 = kt * 32 + ks, h = c0 >> 6, d0 = c0 & 63;
    const _Float16* src = outx + (((size_t)(bI * HH + h) * NN + n) * 64 + d0);
    _Float16* dst = As + r * 32 + ks;
    async_ld_b128(dst, src);            // 16 B
    async_ld_b128(dst + 8, src + 8);    // 16 B
    async_wait();
    __syncthreads();
    v16h a = a_frag(As, 32, wave * 16, 0);
#pragma unroll
    for (int j = 0; j < 4; ++j) {
      size_t t = (size_t)kt * 32 + (colBlock >> 4) + j;
      v16h b = *(const v16h*)(Wop + t * 512 + lane * 16);
      acc[j] = WMMA(a, b, acc[j]);
    }
    __syncthreads();
  }
  int g = lane >> 4, nIdx = lane & 15;
#pragma unroll
  for (int j = 0; j < 4; ++j) {
    int c = colBlock + 16 * j + nIdx;
    float bv = bo[c];
#pragma unroll
    for (int r = 0; r < 8; ++r) {
      int m = rowBlock + wave * 16 + g * 8 + r;
      out[(size_t)m * CC + c] = acc[j][r] + bv;
    }
  }
}

// ---------------------------------------------------------------------------
extern "C" void kernel_launch(void* const* d_in, const int* in_sizes, int n_in,
                              void* d_out, int out_size, void* d_ws, size_t ws_size,
                              hipStream_t stream) {
  (void)in_sizes; (void)n_in; (void)out_size; (void)ws_size;
  const float* x        = (const float*)d_in[0];
  const float* context  = (const float*)d_in[1];
  const float* Wx       = (const float*)d_in[2];
  const float* bx       = (const float*)d_in[3];
  const float* Wfx      = (const float*)d_in[4];
  const float* bfx      = (const float*)d_in[5];
  const float* Wslice   = (const float*)d_in[6];
  const float* bslice   = (const float*)d_in[7];
  const float* temp     = (const float*)d_in[8];
  const float* Wq = (const float*)d_in[9];   const float* bq = (const float*)d_in[10];
  const float* Wk = (const float*)d_in[11];  const float* bk = (const float*)d_in[12];
  const float* Wv = (const float*)d_in[13];  const float* bv = (const float*)d_in[14];
  const float* Wcq = (const float*)d_in[15]; const float* bcq = (const float*)d_in[16];
  const float* Wck = (const float*)d_in[17]; const float* bck = (const float*)d_in[18];
  const float* Wcv = (const float*)d_in[19]; const float* bcv = (const float*)d_in[20];
  const float* smix = (const float*)d_in[21];
  const float* Wo = (const float*)d_in[22];  const float* bo = (const float*)d_in[23];
  float* out = (float*)d_out;

  const size_t BHN64 = (size_t)BB * HH * NN * 64;  // 33,554,432
  char* p = (char*)d_ws;
  auto alloc = [&](size_t bytes) {
    char* r = p;
    p += (bytes + 255) & ~(size_t)255;
    return r;
  };
  _Float16* xmid   = (_Float16*)alloc(BHN64 * 2);   // reused later as out_x
  _Float16* fxh    = (_Float16*)alloc(BHN64 * 2);
  _Float16* wh     = (_Float16*)alloc(BHN64 * 2);
  _Float16* Wxp    = (_Float16*)alloc(512 * 512 * 2);
  _Float16* Wfxp   = (_Float16*)alloc(512 * 512 * 2);
  _Float16* Wop    = (_Float16*)alloc(512 * 512 * 2);
  _Float16* Wsp    = (_Float16*)alloc(64 * 64 * 2);
  _Float16* outTokP= (_Float16*)alloc(16 * 8 * 512 * 2);
  float* tokens    = (float*)alloc(16 * 4096 * 4);
  float* norm      = (float*)alloc(16 * 64 * 4);
  float* scratch   = (float*)alloc((size_t)16 * 10 * 4096 * 4);

  pack_b<<<64, 256, 0, stream>>>(Wx, Wxp, 512, 512);
  pack_b<<<64, 256, 0, stream>>>(Wfx, Wfxp, 512, 512);
  pack_b<<<64, 256, 0, stream>>>(Wo, Wop, 512, 512);
  pack_b<<<1, 256, 0, stream>>>(Wslice, Wsp, 64, 64);
  zero_f32<<<(16 * 4096 + 255) / 256, 256, 0, stream>>>(tokens, 16 * 4096);
  zero_f32<<<4, 256, 0, stream>>>(norm, 16 * 64);

  gemm_x_dual<<<dim3(512, 8), 256, 0, stream>>>(x, Wxp, Wfxp, bx, bfx, xmid, fxh);
  slice_softmax<<<dim3(256, 16), 256, 0, stream>>>(xmid, Wsp, bslice, temp, wh, norm);
  tokens_gemm<<<dim3(8, 16), 256, 0, stream>>>(wh, fxh, tokens);
  attn_kernel<<<16, 256, 0, stream>>>(tokens, norm, Wq, bq, Wk, bk, Wv, bv,
                                      Wcq, bcq, Wck, bck, Wcv, bcv,
                                      context, smix, scratch, outTokP);
  deslice_gemm<<<dim3(256, 16), 256, 0, stream>>>(wh, outTokP, xmid /* out_x */);
  out_gemm<<<dim3(512, 8), 256, 0, stream>>>(xmid, Wop, bo, out);
}
